// GraphEmbedderKBGAT_13786845020226
// MI455X (gfx1250) — matrix operand order repaired and checked by reading, not compiled, not run on hardware
//
#include <hip/hip_runtime.h>
#include <hip/hip_bf16.h>
#include <math.h>

// ---------------- problem constants (match reference) ----------------
#define N_ENT  50000
#define N_RELS 200
#define DIM    128
#define E_FULL 300000
#define E_BATCH 80000
#define NEG_SLOPE 0.2f

typedef __attribute__((ext_vector_type(16))) _Float16 v16h;
typedef __attribute__((ext_vector_type(8)))  float    v8f;

static __device__ __forceinline__ float leaky(float v) {
    return v >= 0.f ? v : NEG_SLOPE * v;
}
// order-preserving float<->uint encoding for atomicMax on floats
static __device__ __forceinline__ unsigned fenc(float f) {
    unsigned u = __float_as_uint(f);
    return (u & 0x80000000u) ? ~u : (u | 0x80000000u);
}
static __device__ __forceinline__ float fdec(unsigned e) {
    return (e & 0x80000000u) ? __uint_as_float(e & 0x7FFFFFFFu)
                             : __uint_as_float(~e);
}

// ---------------------------------------------------------------------
// WMMA f16 GEMM: C[M,N] = A[M,K] @ B   (B row-major [K,N], or [N,K] if TRANSB)
// One wave computes one 16x16 tile with v_wmma_f32_16x16x32_f16.
// Fragment layouts (wave32; A/B use this code's contiguous k order, not the ISA table order):
//   A (16x32 f16): lane l (half=l>>4, row=l&15); slot i -> k = 16*half+i
//                  => one contiguous 16-float group.
//   B (32x16 f16): lane l (col=l&15); slot i -> k = 16*half+i.
//   C/D (16x16 f32): vgpr j -> row j+8*half, col l&15.
// Out-of-range rows/cols are CLAMPED (loads stay unconditional; stores are
// masked), and the K tail (only K=200 call) is one branchless masked step.
// ---------------------------------------------------------------------
template <bool TRANSB>
__global__ void gemm_wmma_f16(const float* __restrict__ A,
                              const float* __restrict__ B,
                              float* __restrict__ C,
                              int M, int N, int K) {
    int waveId = blockIdx.x * (blockDim.x >> 5) + (threadIdx.x >> 5);
    int lane   = threadIdx.x & 31;
    int tilesN = (N + 15) >> 4;
    int tilesM = (M + 15) >> 4;
    int tM = waveId / tilesN;
    int tN = waveId % tilesN;
    if (tM >= tilesM) return;

    const int row0 = tM << 4, col0 = tN << 4;
    const int half = lane >> 4;
    int rA = row0 + (lane & 15);
    int nB = col0 + (lane & 15);
    const bool nBok = (nB < N);
    if (rA >= M) rA = M - 1;          // clamped: garbage rows computed, never stored
    if (nB >= N) nB = N - 1;

    const float* Arow = A + (size_t)rA * K + 16 * half;     // + k0 per step
    const float* Bt   = TRANSB ? (B + (size_t)nB * K + 16 * half)   // contiguous
                               : (B + (size_t)(16 * half) * N + nB); // row-strided

    v8f acc = {};
    const int Kbody = K & ~31;
    int k0 = 0;
    for (; k0 < Kbody; k0 += 32) {
        float av[16], bv[16];
        // A: 16 contiguous floats -> 4x float4 (16B aligned)
        {
            const float4* p0 = (const float4*)(Arow + k0);
            *(float4*)&av[0]  = p0[0];
            *(float4*)&av[4]  = p0[1];
            *(float4*)&av[8]  = p0[2];
            *(float4*)&av[12] = p0[3];
        }
        if (TRANSB) {
            const float4* p = (const float4*)(Bt + k0);
            *(float4*)&bv[0]  = p[0];
            *(float4*)&bv[4]  = p[1];
            *(float4*)&bv[8]  = p[2];
            *(float4*)&bv[12] = p[3];
        } else {
#pragma unroll
            for (int i = 0; i < 16; ++i) bv[i] = Bt[(size_t)(k0 + i) * N];
        }
        v16h a, b;
#pragma unroll
        for (int i = 0; i < 16; ++i) { a[i] = (_Float16)av[i]; b[i] = (_Float16)bv[i]; }
        acc = __builtin_amdgcn_wmma_f32_16x16x32_f16(
            false, a, false, b, (short)0, acc, false, false);
    }
    if (k0 < K) {   // masked tail (K=200 case): clamped loads + branchless zeroing
        v16h a, b;
#pragma unroll
        for (int i = 0; i < 16; ++i) {
            int kk = k0 + 16 * half + i;
            int kc = kk < K ? kk : K - 1;
            float v = A[(size_t)rA * K + kc];
            a[i] = (_Float16)(kk < K ? v : 0.0f);
        }
#pragma unroll
        for (int i = 0; i < 16; ++i) {
            int kk = k0 + 16 * half + i;
            int kc = kk < K ? kk : K - 1;
            float v = TRANSB ? B[(size_t)nB * K + kc] : B[(size_t)kc * N + nB];
            b[i] = (_Float16)(kk < K ? v : 0.0f);
        }
        acc = __builtin_amdgcn_wmma_f32_16x16x32_f16(
            false, a, false, b, (short)0, acc, false, false);
    }
#pragma unroll
    for (int j = 0; j < 8; ++j) {
        int r = row0 + j + 8 * half;
        if (r < M && nBok) C[(size_t)r * N + nB] = acc[j];
    }
}

// per-relation attention-edge scalar: ae[r*H+h] = dot(etr[r, h*C:(h+1)*C], att_e[h])
__global__ void k_ae_rel(const float* __restrict__ etr, const float* __restrict__ att_e,
                         float* __restrict__ ae, int R, int H, int Cc) {
    int t = blockIdx.x * blockDim.x + threadIdx.x;
    if (t >= R * H) return;
    int r = t / H, h = t % H;
    const float* er = etr + (size_t)r * H * Cc + (size_t)h * Cc;
    const float* av = att_e + (size_t)h * Cc;
    float s = 0.f;
    for (int c = 0; c < Cc; ++c) s += er[c] * av[c];
    ae[t] = s;
}

// deg + self-loop edge-attr scalar accumulation (both layers at once)
__global__ void k_deg_aeloop(const int* __restrict__ rel, const int* __restrict__ dst,
                             const float* __restrict__ ae1, const float* __restrict__ ae2,
                             float* __restrict__ deg,
                             float* __restrict__ aeloop1, float* __restrict__ aeloop2,
                             int E) {
    int e = blockIdx.x * blockDim.x + threadIdx.x;
    if (e >= E) return;
    int d = dst[e], r = rel[e];
    atomicAdd(&deg[d], 1.0f);
    atomicAdd(&aeloop1[d * 2 + 0], ae1[r * 2 + 0]);
    atomicAdd(&aeloop1[d * 2 + 1], ae1[r * 2 + 1]);
    atomicAdd(&aeloop2[d * 2 + 0], ae2[r * 2 + 0]);
    atomicAdd(&aeloop2[d * 2 + 1], ae2[r * 2 + 1]);
}

__global__ void k_aeloop_norm(const float* __restrict__ deg,
                              float* __restrict__ aeloop1, float* __restrict__ aeloop2,
                              int Nn) {
    int n = blockIdx.x * blockDim.x + threadIdx.x;
    if (n >= Nn) return;
    float inv = 1.0f / fmaxf(deg[n], 1.0f);
    aeloop1[n * 2 + 0] *= inv; aeloop1[n * 2 + 1] *= inv;
    aeloop2[n * 2 + 0] *= inv; aeloop2[n * 2 + 1] *= inv;
}

// per-node attention scalars: a_s[n*H+h], a_d[n*H+h]
__global__ void k_attdot(const float* __restrict__ xt,
                         const float* __restrict__ att_s, const float* __restrict__ att_d,
                         float* __restrict__ a_s, float* __restrict__ a_d,
                         int Nn, int H, int Cc) {
    int t = blockIdx.x * blockDim.x + threadIdx.x;
    if (t >= Nn * H) return;
    int n = t / H, h = t % H;
    const float* xr = xt + (size_t)n * H * Cc + (size_t)h * Cc;
    const float* as = att_s + (size_t)h * Cc;
    const float* ad = att_d + (size_t)h * Cc;
    float ss = 0.f, sd = 0.f;
    for (int c = 0; c < Cc; c += 4) {
        float4 v = *(const float4*)(xr + c);
        float4 s4 = *(const float4*)(as + c);
        float4 d4 = *(const float4*)(ad + c);
        ss += v.x * s4.x + v.y * s4.y + v.z * s4.z + v.w * s4.w;
        sd += v.x * d4.x + v.y * d4.y + v.z * d4.z + v.w * d4.w;
    }
    a_s[t] = ss; a_d[t] = sd;
}

// softmax pass 1: segment max of leaky(logit) over dst (incl. self loops at i>=E)
__global__ void k_pass1(const int* __restrict__ src, const int* __restrict__ rel,
                        const int* __restrict__ dst, int E, int Nn,
                        const float* __restrict__ a_s, const float* __restrict__ a_d,
                        const float* __restrict__ ae_rel, const float* __restrict__ aeloop,
                        unsigned* __restrict__ amax) {
    int i = blockIdx.x * blockDim.x + threadIdx.x;
    if (i >= E + Nn) return;
    int s, d; float e0, e1;
    if (i < E) { s = src[i]; d = dst[i]; int r = rel[i]; e0 = ae_rel[r * 2]; e1 = ae_rel[r * 2 + 1]; }
    else       { s = d = i - E; e0 = aeloop[d * 2]; e1 = aeloop[d * 2 + 1]; }
    float al0 = leaky(a_s[s * 2 + 0] + a_d[d * 2 + 0] + e0);
    float al1 = leaky(a_s[s * 2 + 1] + a_d[d * 2 + 1] + e1);
    atomicMax(&amax[d * 2 + 0], fenc(al0));
    atomicMax(&amax[d * 2 + 1], fenc(al1));
}

// softmax pass 2: ex = exp(logit - max); store per-edge, accumulate denominator
__global__ void k_pass2(const int* __restrict__ src, const int* __restrict__ rel,
                        const int* __restrict__ dst, int E, int Nn,
                        const float* __restrict__ a_s, const float* __restrict__ a_d,
                        const float* __restrict__ ae_rel, const float* __restrict__ aeloop,
                        const unsigned* __restrict__ amax,
                        float* __restrict__ exbuf, float* __restrict__ den) {
    int i = blockIdx.x * blockDim.x + threadIdx.x;
    if (i >= E + Nn) return;
    int s, d; float e0, e1;
    if (i < E) { s = src[i]; d = dst[i]; int r = rel[i]; e0 = ae_rel[r * 2]; e1 = ae_rel[r * 2 + 1]; }
    else       { s = d = i - E; e0 = aeloop[d * 2]; e1 = aeloop[d * 2 + 1]; }
    float al0 = leaky(a_s[s * 2 + 0] + a_d[d * 2 + 0] + e0);
    float al1 = leaky(a_s[s * 2 + 1] + a_d[d * 2 + 1] + e1);
    float x0 = __expf(al0 - fdec(amax[d * 2 + 0]));
    float x1 = __expf(al1 - fdec(amax[d * 2 + 1]));
    exbuf[i * 2 + 0] = x0; exbuf[i * 2 + 1] = x1;
    atomicAdd(&den[d * 2 + 0], x0);
    atomicAdd(&den[d * 2 + 1], x1);
}

// layer-1 aggregation: wave per edge, lane per channel, HC=128 (H=2,C=64)
__global__ void k_agg_l1(const int* __restrict__ src, const int* __restrict__ dst,
                         int E, int Nn,
                         const float* __restrict__ xt, const float* __restrict__ exbuf,
                         const float* __restrict__ den, float* __restrict__ out) {
    int wid  = blockIdx.x * (blockDim.x >> 5) + (threadIdx.x >> 5);
    int lane = threadIdx.x & 31;
    if (wid >= E + Nn) return;
    int s, d;
    if (wid < E) { s = src[wid]; d = dst[wid]; } else { s = d = wid - E; }
    float w0 = exbuf[wid * 2 + 0] / (den[d * 2 + 0] + 1e-16f);
    float w1 = exbuf[wid * 2 + 1] / (den[d * 2 + 1] + 1e-16f);
    const float* xr = xt + (size_t)s * 128;
    float* orow = out + (size_t)d * 128;
#pragma unroll
    for (int j = 0; j < 4; ++j) {
        int c = lane + 32 * j;
        float w = (c < 64) ? w0 : w1;
        atomicAdd(&orow[c], xr[c] * w);
    }
}

// layer-2 aggregation with fused head-mean: out[d,c] += 0.5*(w0*xt2[s,c]+w1*xt2[s,128+c])
__global__ void k_agg_l2(const int* __restrict__ src, const int* __restrict__ dst,
                         int E, int Nn,
                         const float* __restrict__ xt2, const float* __restrict__ exbuf,
                         const float* __restrict__ den, float* __restrict__ out) {
    int wid  = blockIdx.x * (blockDim.x >> 5) + (threadIdx.x >> 5);
    int lane = threadIdx.x & 31;
    if (wid >= E + Nn) return;
    int s, d;
    if (wid < E) { s = src[wid]; d = dst[wid]; } else { s = d = wid - E; }
    float w0 = 0.5f * exbuf[wid * 2 + 0] / (den[d * 2 + 0] + 1e-16f);
    float w1 = 0.5f * exbuf[wid * 2 + 1] / (den[d * 2 + 1] + 1e-16f);
    const float* xr = xt2 + (size_t)s * 256;
    float* orow = out + (size_t)d * 128;
#pragma unroll
    for (int j = 0; j < 4; ++j) {
        int c = lane + 32 * j;
        atomicAdd(&orow[c], xr[c] * w0 + xr[128 + c] * w1);
    }
}

__global__ void k_bias_leaky(float* __restrict__ x, const float* __restrict__ b, int total) {
    int t = blockIdx.x * blockDim.x + threadIdx.x;
    if (t >= total) return;
    x[t] = leaky(x[t] + b[t & 127]);
}

// xm[n] = normalize( xm[n] + 0.5*(b2[c]+b2[c+128]) + skip[n] ); wave per node
__global__ void k_combine_norm(float* __restrict__ xm, const float* __restrict__ skip,
                               const float* __restrict__ b2, int Nn) {
    int n    = blockIdx.x * (blockDim.x >> 5) + (threadIdx.x >> 5);
    int lane = threadIdx.x & 31;
    if (n >= Nn) return;
    float v[4]; float ss = 0.f;
#pragma unroll
    for (int j = 0; j < 4; ++j) {
        int c = lane + 32 * j;
        float t = xm[(size_t)n * 128 + c] + 0.5f * (b2[c] + b2[128 + c])
                + skip[(size_t)n * 128 + c];
        v[j] = t; ss += t * t;
    }
#pragma unroll
    for (int m = 16; m >= 1; m >>= 1) ss += __shfl_xor(ss, m, 32);
    float sc = 1.0f / fmaxf(sqrtf(ss), 1e-12f);
#pragma unroll
    for (int j = 0; j < 4; ++j) {
        int c = lane + 32 * j;
        xm[(size_t)n * 128 + c] = v[j] * sc;
    }
}

// edge_emb gather (float4 vectorized): out[0:EB*128)=x[idx0], then x[idx1]
__global__ void k_gather(const int* __restrict__ idx0, const int* __restrict__ idx1,
                         const float* __restrict__ x, float* __restrict__ out, int EB) {
    int t = blockIdx.x * blockDim.x + threadIdx.x;       // over EB*32 float4s
    if (t >= EB * 32) return;
    int e = t >> 5, c4 = t & 31;
    const float4* xr0 = (const float4*)(x + (size_t)idx0[e] * 128);
    const float4* xr1 = (const float4*)(x + (size_t)idx1[e] * 128);
    ((float4*)out)[t]                    = xr0[c4];
    ((float4*)out)[(size_t)EB * 32 + t]  = xr1[c4];
}

static inline int cdiv(long long a, long long b) { return (int)((a + b - 1) / b); }

extern "C" void kernel_launch(void* const* d_in, const int* in_sizes, int n_in,
                              void* d_out, int out_size, void* d_ws, size_t ws_size,
                              hipStream_t stream) {
    const int*   edge_index = (const int*)  d_in[0];
    const float* edge_attr  = (const float*)d_in[1];
    const int*   edge_data  = (const int*)  d_in[2];
    const float* ent_emb    = (const float*)d_in[3];
    const float* rel_emb    = (const float*)d_in[4];
    const float* W1    = (const float*)d_in[5];
    const float* We1   = (const float*)d_in[6];
    const float* att_s1= (const float*)d_in[7];
    const float* att_d1= (const float*)d_in[8];
    const float* att_e1= (const float*)d_in[9];
    const float* b1    = (const float*)d_in[10];
    const float* W2    = (const float*)d_in[11];
    const float* We2   = (const float*)d_in[12];
    const float* att_s2= (const float*)d_in[13];
    const float* att_d2= (const float*)d_in[14];
    const float* att_e2= (const float*)d_in[15];
    const float* b2    = (const float*)d_in[16];
    const float* W_skip= (const float*)d_in[17];
    const float* W_r   = (const float*)d_in[18];
    float* out = (float*)d_out;

    const int N = N_ENT, E = E_FULL, EB = E_BATCH, EN = E_FULL + N_ENT;
    const int* src = edge_data;
    const int* rel = edge_data + E;
    const int* dst = edge_data + 2 * E;
    const int* idx0 = edge_index;
    const int* idx1 = edge_index + EB;

    // ---------------- workspace arena (floats; all offsets 16-float aligned) --
    float* ws = (float*)d_ws;
    size_t o = 0;
    float*    deg     = ws + o; o += N;        // |-- zeroed block start
    float*    aeloop1 = ws + o; o += 2 * N;
    float*    aeloop2 = ws + o; o += 2 * N;
    unsigned* amax1   = (unsigned*)(ws + o); o += 2 * N;
    float*    den1    = ws + o; o += 2 * N;
    unsigned* amax2   = (unsigned*)(ws + o); o += 2 * N;
    float*    den2    = ws + o; o += 2 * N;    // |-- zeroed block end (13N)
    float*    a_s1    = ws + o; o += 2 * N;
    float*    a_d1    = ws + o; o += 2 * N;
    float*    a_s2    = ws + o; o += 2 * N;
    float*    a_d2    = ws + o; o += 2 * N;
    float*    ex1     = ws + o; o += 2 * (size_t)EN;
    float*    ex2     = ws + o; o += 2 * (size_t)EN;
    float*    etr1    = ws + o; o += (size_t)N_RELS * 128;
    float*    etr2    = ws + o; o += (size_t)N_RELS * 256;
    float*    ae1     = ws + o; o += N_RELS * 2;
    float*    ae2     = ws + o; o += N_RELS * 2;
    float*    bufA    = ws + o; o += (size_t)N * 128;  // xt1, then layer-2 mean-agg / final x
    float*    bufB    = ws + o; o += (size_t)N * 128;  // out1/x1, then skip
    float*    bufC    = ws + o; o += (size_t)N * 256;  // xt2
    (void)ws_size; (void)n_in; (void)in_sizes; (void)out_size;

    // zero accumulators (deg..den2 is contiguous)
    hipMemsetAsync(deg, 0, (size_t)13 * N * sizeof(float), stream);

    const int TB = 256;
    const int WPB = TB / 32;

    // relation-side GEMMs (small, still WMMA): etr = rel_emb @ We
    gemm_wmma_f16<false><<<cdiv((long long)cdiv(N_RELS,16) * (128/16), WPB), TB, 0, stream>>>(
        rel_emb, We1, etr1, N_RELS, 128, 128);
    gemm_wmma_f16<false><<<cdiv((long long)cdiv(N_RELS,16) * (256/16), WPB), TB, 0, stream>>>(
        rel_emb, We2, etr2, N_RELS, 256, 128);
    k_ae_rel<<<cdiv(N_RELS * 2, TB), TB, 0, stream>>>(etr1, att_e1, ae1, N_RELS, 2, 64);
    k_ae_rel<<<cdiv(N_RELS * 2, TB), TB, 0, stream>>>(etr2, att_e2, ae2, N_RELS, 2, 128);

    // degree + self-loop edge-attr scalars (fill_value='mean')
    k_deg_aeloop<<<cdiv(E, TB), TB, 0, stream>>>(rel, dst, ae1, ae2, deg, aeloop1, aeloop2, E);
    k_aeloop_norm<<<cdiv(N, TB), TB, 0, stream>>>(deg, aeloop1, aeloop2, N);

    // ---- layer 1 ----
    gemm_wmma_f16<false><<<cdiv((long long)(N/16) * (128/16), WPB), TB, 0, stream>>>(
        ent_emb, W1, bufA, N, 128, 128);                       // xt1
    k_attdot<<<cdiv(N * 2, TB), TB, 0, stream>>>(bufA, att_s1, att_d1, a_s1, a_d1, N, 2, 64);
    k_pass1<<<cdiv(EN, TB), TB, 0, stream>>>(src, rel, dst, E, N, a_s1, a_d1, ae1, aeloop1, amax1);
    k_pass2<<<cdiv(EN, TB), TB, 0, stream>>>(src, rel, dst, E, N, a_s1, a_d1, ae1, aeloop1,
                                             amax1, ex1, den1);
    hipMemsetAsync(bufB, 0, (size_t)N * 128 * sizeof(float), stream);
    k_agg_l1<<<cdiv(EN, WPB), TB, 0, stream>>>(src, dst, E, N, bufA, ex1, den1, bufB);
    k_bias_leaky<<<cdiv((long long)N * 128, TB), TB, 0, stream>>>(bufB, b1, N * 128);  // x1

    // ---- layer 2 ----
    gemm_wmma_f16<false><<<cdiv((long long)(N/16) * (256/16), WPB), TB, 0, stream>>>(
        bufB, W2, bufC, N, 256, 128);                          // xt2
    // skip = ent_emb @ W_skip^T  (x1 no longer needed -> reuse bufB)
    gemm_wmma_f16<true><<<cdiv((long long)(N/16) * (128/16), WPB), TB, 0, stream>>>(
        ent_emb, W_skip, bufB, N, 128, 128);
    k_attdot<<<cdiv(N * 2, TB), TB, 0, stream>>>(bufC, att_s2, att_d2, a_s2, a_d2, N, 2, 128);
    k_pass1<<<cdiv(EN, TB), TB, 0, stream>>>(src, rel, dst, E, N, a_s2, a_d2, ae2, aeloop2, amax2);
    k_pass2<<<cdiv(EN, TB), TB, 0, stream>>>(src, rel, dst, E, N, a_s2, a_d2, ae2, aeloop2,
                                             amax2, ex2, den2);
    hipMemsetAsync(bufA, 0, (size_t)N * 128 * sizeof(float), stream);  // xt1 dead
    k_agg_l2<<<cdiv(EN, WPB), TB, 0, stream>>>(src, dst, E, N, bufC, ex2, den2, bufA);

    // combine heads-mean + bias-mean + skip, then row-normalize (in place in bufA)
    k_combine_norm<<<cdiv(N, WPB), TB, 0, stream>>>(bufA, bufB, b2, N);

    // outputs: edge_emb gather, then edge_attr @ W_r^T via WMMA
    k_gather<<<cdiv((long long)EB * 32, TB), TB, 0, stream>>>(idx0, idx1, bufA, out, EB);
    gemm_wmma_f16<true><<<cdiv((long long)(EB/16) * (128/16), WPB), TB, 0, stream>>>(
        edge_attr, W_r, out + (size_t)2 * EB * 128, EB, 128, N_RELS /*K=200, tail-masked*/);
}
